// code2vec_6949257085255
// MI455X (gfx1250) — matrix operand order repaired and checked by reading, not compiled, not run on hardware
//
#include <hip/hip_runtime.h>
#include <hip/hip_bf16.h>

typedef __attribute__((ext_vector_type(16))) __bf16 v16bf;
typedef __attribute__((ext_vector_type(8)))  __bf16 v8bf;
typedef __attribute__((ext_vector_type(8)))  float  v8f;

#define CTX    200
#define EMB    128
#define KDIM   384          // 3*EMB
#define CODE   50
#define NTILES 13           // ceil(200/16)
#define AROW   392          // padded A row stride (halfwords), 49*8 -> 16B aligned chunks
#define FROW   52           // padded flat row stride (floats)

static __device__ __forceinline__ unsigned short f2bf(float f) {
    unsigned u = __builtin_bit_cast(unsigned, f);
    unsigned r = u + 0x7FFFu + ((u >> 16) & 1u);   // round-to-nearest-even
    return (unsigned short)(r >> 16);
}

// Branch-free tanh: 1 - 2/(exp(2x)+1).  Saturates to +/-1 at large |x|,
// lowers to v_exp_f32 + v_rcp_f32 (no exec-mask branching like libm tanhf).
static __device__ __forceinline__ float tanh_fast(float x) {
    const float e = __expf(2.0f * x);
    return 1.0f - 2.0f * __builtin_amdgcn_rcpf(e + 1.0f);
}

__global__ __launch_bounds__(128)
void code2vec_attn_kernel(const int* __restrict__ e1,
                          const int* __restrict__ p,
                          const int* __restrict__ e2,
                          const float* __restrict__ entsE,
                          const float* __restrict__ pathE,
                          const float* __restrict__ TM,
                          const float* __restrict__ attP,
                          float* __restrict__ out) {
    // Double-buffered bf16 A staging: gather tile ct+1 overlaps WMMA of tile ct.
    __shared__ __align__(32) unsigned short AsS[2][16 * AROW];
    __shared__ __align__(16) float flatS[16 * NTILES * FROW];   // tanh(ctx @ TM), [208][52]
    __shared__ float scoresS[16 * NTILES];
    __shared__ float attS[64];
    __shared__ float redS[8];

    const int b    = blockIdx.x;
    const int tid  = threadIdx.x;
    const int lane = tid & 31;
    const int w    = tid >> 5;        // wave id = N-tile id (cols 16w..16w+15)
    const int half = lane >> 4;       // K-half selector (this code's contiguous k = 16*half + s order)
    const int M    = lane & 15;

    if (tid < CODE) attS[tid] = attP[tid];
    if (tid >= CODE && tid < 64) attS[tid] = 0.0f;

    // ---- Load this wave's B fragments (transform matrix, bf16) into VGPRs.
    // B 32x16 layout: lane (khalf*16 + n) holds 16 consecutive K values of column n.
    const int ncol = w * 16 + M;
    v16bf bfrag[12];
#pragma unroll
    for (int kb = 0; kb < 12; ++kb) {
        union { unsigned short s[16]; v16bf v; } u;
#pragma unroll
        for (int j = 0; j < 16; ++j) {
            const int k = kb * 32 + half * 16 + j;
            const float val = (ncol < CODE) ? TM[k * CODE + ncol] : 0.0f;
            u.s[j] = f2bf(val);
        }
        bfrag[kb] = u.v;
    }

    // ---- Gather one 16-context tile into a staging buffer.
    // Wave w handles rows w*12 .. w*12+11, so seg = i%3 is a compile-time
    // constant after unrolling (no exec-mask branching), and each row is a
    // fully coalesced 512B load per wave.
    auto gather_tile = [&](int ct, unsigned short* __restrict__ buf) {
#pragma unroll
        for (int i = 0; i < 12; ++i) {
            const int seg = i % 3;                 // static after unroll
            const int cl  = w * 4 + i / 3;
            int c = ct * 16 + cl; if (c > CTX - 1) c = CTX - 1;
            const int*   idxp = (seg == 0) ? e1 : (seg == 1) ? p : e2;
            const float* tab  = (seg == 1) ? pathE : entsE;
            const int row = idxp[b * CTX + c];
            const float4 v = ((const float4*)(tab + (size_t)row * EMB))[lane];
            const int o = cl * AROW + seg * EMB + lane * 4;
            buf[o + 0] = f2bf(v.x);
            buf[o + 1] = f2bf(v.y);
            buf[o + 2] = f2bf(v.z);
            buf[o + 3] = f2bf(v.w);
        }
    };

    // Prologue: fill buffer 0 with tile 0.
    gather_tile(0, AsS[0]);
    __syncthreads();

    // ---- Main loop over 13 tiles of 16 contexts (software pipelined).
    for (int ct = 0; ct < NTILES; ++ct) {
        const int cur = ct & 1;
        // Issue next tile's gather into the other buffer (overlaps WMMAs).
        if (ct < NTILES - 1) gather_tile(ct + 1, AsS[cur ^ 1]);

        // WMMA: 12 K-steps of 16x16x32 bf16, f32 accumulate.
        const unsigned short* abuf = AsS[cur];
        v8f acc = {};
#pragma unroll
        for (int kb = 0; kb < 12; ++kb) {
            const int k0 = kb * 32 + half * 16;  // A layout: 16 contiguous K at k0
            const v8bf lo = *(const v8bf*)&abuf[M * AROW + k0];
            const v8bf hi = *(const v8bf*)&abuf[M * AROW + k0 + 8];
            const v16bf a = __builtin_shufflevector(lo, hi,
                0, 1, 2, 3, 4, 5, 6, 7, 8, 9, 10, 11, 12, 13, 14, 15);
            acc = __builtin_amdgcn_wmma_f32_16x16x32_bf16(
                false, a, false, bfrag[kb], (short)0, acc, false, false);
        }

        // tanh + store flat. D layout: VGPR r -> M=r+8*half, N=lane&15.
        const int col = w * 16 + M;
#pragma unroll
        for (int r = 0; r < 8; ++r) {
            const int row = ct * 16 + r + half * 8;
            if (col < CODE) flatS[row * FROW + col] = tanh_fast(acc[r]);
        }
        __syncthreads();   // next tile's buffer fully written; flat visible later
    }

    // ---- Attention scores.
    for (int c = tid; c < CTX; c += 128) {
        const float* fr = &flatS[c * FROW];
        float s = 0.0f;
#pragma unroll 10
        for (int d = 0; d < CODE; ++d) s += fr[d] * attS[d];
        scoresS[c] = s;
    }
    __syncthreads();

    // ---- Softmax (block reduction, wave32 shfl).
    float pm = -1e30f;
    for (int c = tid; c < CTX; c += 128) pm = fmaxf(pm, scoresS[c]);
#pragma unroll
    for (int off = 16; off > 0; off >>= 1) pm = fmaxf(pm, __shfl_xor(pm, off, 32));
    if (lane == 0) redS[w] = pm;
    __syncthreads();
    const float bmax = fmaxf(fmaxf(redS[0], redS[1]), fmaxf(redS[2], redS[3]));

    float ps = 0.0f;
    for (int c = tid; c < CTX; c += 128) {
        const float e = __expf(scoresS[c] - bmax);
        scoresS[c] = e;
        ps += e;
    }
#pragma unroll
    for (int off = 16; off > 0; off >>= 1) ps += __shfl_xor(ps, off, 32);
    if (lane == 0) redS[4 + w] = ps;
    __syncthreads();
    const float inv = 1.0f / (redS[4] + redS[5] + redS[6] + redS[7]);

    // ---- Weighted-sum pooling.
    for (int d = tid; d < CODE; d += 128) {
        float a0 = 0.0f;
        for (int c = 0; c < CTX; ++c) a0 += flatS[c * FROW + d] * scoresS[c];
        out[b * CODE + d] = a0 * inv;
    }
}

extern "C" void kernel_launch(void* const* d_in, const int* in_sizes, int n_in,
                              void* d_out, int out_size, void* d_ws, size_t ws_size,
                              hipStream_t stream) {
    const int*   e1    = (const int*)d_in[0];
    const int*   p     = (const int*)d_in[1];
    const int*   e2    = (const int*)d_in[2];
    // d_in[3] = tags (unused by forward output)
    const float* entsE = (const float*)d_in[4];
    const float* pathE = (const float*)d_in[5];
    // d_in[6] = tags_embeddings (unused by forward output)
    const float* TM    = (const float*)d_in[7];
    const float* attP  = (const float*)d_in[8];
    float* out = (float*)d_out;

    const int batch = in_sizes[3];   // 1024
    code2vec_attn_kernel<<<dim3(batch), dim3(128), 0, stream>>>(
        e1, p, e2, entsE, pathE, TM, attP, out);
}
